// PointActorRecurrent_52295521796462
// MI455X (gfx1250) — matrix operand order repaired and checked by reading, not compiled, not run on hardware
//
#include <hip/hip_runtime.h>

// ---------------------------------------------------------------------------
// PointActorRecurrent on gfx1250 (CDNA5, wave32, WMMA 16x16x32 f16->f32)
//   B=256, N=1024, D_IN=3, HID=256, ACT=6, T=50
// ---------------------------------------------------------------------------

#define BATCH 256
#define NPTS  1024
#define HID   256
#define ACT   6
#define TSTEPS 50

typedef __attribute__((ext_vector_type(16))) _Float16 v16h;
typedef __attribute__((ext_vector_type(8)))  float    v8f;

// ---- WMMA fragment helpers (layouts per CDNA5 ISA §7.12.2) ------

// A-matrix 16x32 f16, row-major source with leading dim lda.
__device__ __forceinline__ v16h frag_a(const _Float16* A, int lda, int m0, int k0, int lane) {
  const _Float16* p = A + (m0 + (lane & 15)) * lda + k0 + ((lane >> 4) << 3);
  v16h r;
#pragma unroll
  for (int i = 0; i < 8; ++i) r[i] = p[i];
#pragma unroll
  for (int i = 0; i < 8; ++i) r[8 + i] = p[16 + i];
  return r;
}

// B-matrix 32x16 f16 from an N-major ("Bt") buffer: Bt[n*ldb + k] == B[k][n].
__device__ __forceinline__ v16h frag_b(const _Float16* Bt, int ldb, int n0, int k0, int lane) {
  const _Float16* p = Bt + (n0 + (lane & 15)) * ldb + k0 + ((lane >> 4) << 3);
  v16h r;
#pragma unroll
  for (int i = 0; i < 8; ++i) r[i] = p[i];
#pragma unroll
  for (int i = 0; i < 8; ++i) r[8 + i] = p[16 + i];
  return r;
}

__device__ __forceinline__ v8f wmma16(v16h a, v16h b, v8f c) {
  return __builtin_amdgcn_wmma_f32_16x16x32_f16(false, a, false, b, (short)0, c, false, false);
}

// ---- gfx1250 async memory->LDS copy (CDNA5 ISA) ------
// LDS dest = LDS base + vdst VGPR value; LDS aperture flat addr[31:0] is the
// wave-relative LDS byte offset, so truncating a generic shared pointer works.
__device__ __forceinline__ void async_copy_b128(unsigned lds_off, const void* gsrc) {
  asm volatile("global_load_async_to_lds_b128 %0, %1, off"
               :: "v"(lds_off), "v"(gsrc)
               : "memory");
}
__device__ __forceinline__ void wait_asynccnt0() {
  asm volatile("s_wait_asynccnt 0x0" ::: "memory");
}

// ---- weight conversion: f32 [K,N] -> f16 N-major [N][K] --------------------
__global__ void convT_kernel(const float* __restrict__ in, _Float16* __restrict__ out,
                             int K, int N) {
  int i = blockIdx.x * blockDim.x + threadIdx.x;
  if (i < K * N) {
    int n = i / K, k = i - n * K;
    out[i] = (_Float16)in[k * N + n];
  }
}

__global__ void convC_kernel(const float* __restrict__ in, _Float16* __restrict__ out, int n) {
  int i = blockIdx.x * blockDim.x + threadIdx.x;
  if (i < n) out[i] = (_Float16)in[i];
}

// ---- encoder: per-point MLP 3->64->128->512 (relu) + max-pool --------------
// grid = 2048 blocks x 256 threads; each block: 128 points of one batch elem.
__global__ __launch_bounds__(256) void enc_kernel(
    const float* __restrict__ data,
    const float* __restrict__ w1, const float* __restrict__ b1,
    const _Float16* __restrict__ w2t, const float* __restrict__ b2,
    const _Float16* __restrict__ w3t, const float* __restrict__ b3,
    int* __restrict__ pooled /* [BATCH][512] float bits, pre-zeroed */) {
  __shared__ _Float16 act1[128][64];
  __shared__ _Float16 act2[128][128];
  __shared__ _Float16 w3s[2][16 * 128];   // double-buffered B tile (4KB each)
  __shared__ int pool[512];
  __shared__ float w1s[192];
  __shared__ float b1s[64];

  const int tid  = threadIdx.x;
  const int lane = tid & 31;
  const int wave = tid >> 5;
  const long row0 = (long)blockIdx.x * 128;
  const int  bIdx = blockIdx.x >> 3;   // 8 tiles of 128 points per batch elem

  pool[tid] = 0; pool[tid + 256] = 0;
  if (tid < 192) w1s[tid] = w1[tid];
  if (tid < 64)  b1s[tid] = b1[tid];
  __syncthreads();

  // Layer 1 (K=3): VALU, 2 threads per point row
  {
    int r  = tid >> 1;
    int c0 = (tid & 1) * 32;
    const float* xp = data + (row0 + r) * 3;
    float x0 = xp[0], x1 = xp[1], x2 = xp[2];
#pragma unroll
    for (int c = 0; c < 32; ++c) {
      int cc = c0 + c;
      float v = b1s[cc] + x0 * w1s[cc] + x1 * w1s[64 + cc] + x2 * w1s[128 + cc];
      act1[r][cc] = (_Float16)fmaxf(v, 0.f);
    }
  }
  __syncthreads();

  // Layer 2: [128,64] @ [64,128] -- each wave owns a 16-row stripe
  {
    int m0 = wave * 16;
    for (int nt = 0; nt < 8; ++nt) {
      v8f acc = {};
#pragma unroll
      for (int kt = 0; kt < 2; ++kt) {
        v16h a  = frag_a(&act1[0][0], 64, m0, kt * 32, lane);
        v16h bm = frag_b(w2t, 64, nt * 16, kt * 32, lane);
        acc = wmma16(a, bm, acc);
      }
      int n  = nt * 16 + (lane & 15);
      int rb = m0 + ((lane >> 4) << 3);
      float bias = b2[n];
#pragma unroll
      for (int i = 0; i < 8; ++i)
        act2[rb + i][n] = (_Float16)fmaxf(acc[i] + bias, 0.f);
    }
  }

  // Prologue: async-stage B tile nt=0 (16 n-rows x 128 K halves = 4KB)
  // 256 threads x 16B. Overlaps with nothing yet; waited below with the
  // act2 barrier.
  {
    unsigned dst = (unsigned)(size_t)(&w3s[0][0]) + (unsigned)tid * 16u;
    const char* src = (const char*)w3t + tid * 16;
    async_copy_b128(dst, src);
  }
  wait_asynccnt0();
  __syncthreads();   // act2 ready + w3s[0] ready for all waves

  // Layer 3: [128,128] @ [128,512] + fused relu-max-pool.
  // B tiles staged through LDS (shared by all 8 waves), double-buffered with
  // async copies overlapping the 32 WMMAs of the previous tile.
  {
    int m0 = wave * 16;
    for (int nt = 0; nt < 32; ++nt) {
      int cur = nt & 1;
      if (nt + 1 < 32) {
        unsigned dst = (unsigned)(size_t)(&w3s[cur ^ 1][0]) + (unsigned)tid * 16u;
        const char* src = (const char*)(w3t + (nt + 1) * 16 * 128) + tid * 16;
        async_copy_b128(dst, src);
      }
      v8f acc = {};
#pragma unroll
      for (int kt = 0; kt < 4; ++kt) {
        v16h a  = frag_a(&act2[0][0], 128, m0, kt * 32, lane);
        v16h bm = frag_b(&w3s[cur][0], 128, 0, kt * 32, lane);
        acc = wmma16(a, bm, acc);
      }
      int n = nt * 16 + (lane & 15);
      float bias = b3[n];
      float m = 0.f;                       // relu identity == pool identity
#pragma unroll
      for (int i = 0; i < 8; ++i) m = fmaxf(m, acc[i] + bias);
      m = fmaxf(m, __shfl_xor(m, 16, 32)); // rows 0-7 vs 8-15 of the stripe
      if (lane < 16) atomicMax(&pool[n], __float_as_int(m));
      wait_asynccnt0();    // next tile landed
      __syncthreads();     // all waves done with cur before it is re-staged
    }
  }

  atomicMax(&pooled[bIdx * 512 + tid],       pool[tid]);
  atomicMax(&pooled[bIdx * 512 + 256 + tid], pool[tid + 256]);
}

// ---- head MLP 512->256->128->HID, produces GRU h0 (f32 + f16 copies) ------
__global__ __launch_bounds__(256) void head_kernel(
    const int* __restrict__ pooled_bits,
    const _Float16* __restrict__ m1t, const float* __restrict__ mb1,
    const _Float16* __restrict__ m2t, const float* __restrict__ mb2,
    const _Float16* __restrict__ m3t, const float* __restrict__ mb3,
    float* __restrict__ h0, _Float16* __restrict__ h016) {
  __shared__ _Float16 s0[16][512];
  __shared__ _Float16 s1[16][256];
  __shared__ _Float16 s2[16][128];
  const int tid = threadIdx.x, lane = tid & 31, wave = tid >> 5;
  const int brow0 = blockIdx.x * 16;

  const int* pp = pooled_bits + brow0 * 512;
  for (int i = tid; i < 16 * 512; i += 256)
    s0[i >> 9][i & 511] = (_Float16)__int_as_float(pp[i]);
  __syncthreads();

  // L1: N=256, K=512 ; 16 n-tiles -> 2 per wave
  for (int q = 0; q < 2; ++q) {
    int nt = wave * 2 + q;
    v8f acc = {};
#pragma unroll
    for (int kt = 0; kt < 16; ++kt) {
      v16h a = frag_a(&s0[0][0], 512, 0, kt * 32, lane);
      v16h b = frag_b(m1t, 512, nt * 16, kt * 32, lane);
      acc = wmma16(a, b, acc);
    }
    int n = nt * 16 + (lane & 15), rb = (lane >> 4) << 3;
    float bias = mb1[n];
#pragma unroll
    for (int i = 0; i < 8; ++i) s1[rb + i][n] = (_Float16)fmaxf(acc[i] + bias, 0.f);
  }
  __syncthreads();

  // L2: N=128, K=256 ; 8 n-tiles -> 1 per wave
  {
    int nt = wave;
    v8f acc = {};
#pragma unroll
    for (int kt = 0; kt < 8; ++kt) {
      v16h a = frag_a(&s1[0][0], 256, 0, kt * 32, lane);
      v16h b = frag_b(m2t, 256, nt * 16, kt * 32, lane);
      acc = wmma16(a, b, acc);
    }
    int n = nt * 16 + (lane & 15), rb = (lane >> 4) << 3;
    float bias = mb2[n];
#pragma unroll
    for (int i = 0; i < 8; ++i) s2[rb + i][n] = (_Float16)fmaxf(acc[i] + bias, 0.f);
  }
  __syncthreads();

  // L3: N=HID=256, K=128 ; no relu
  for (int q = 0; q < 2; ++q) {
    int nt = wave * 2 + q;
    v8f acc = {};
#pragma unroll
    for (int kt = 0; kt < 4; ++kt) {
      v16h a = frag_a(&s2[0][0], 128, 0, kt * 32, lane);
      v16h b = frag_b(m3t, 128, nt * 16, kt * 32, lane);
      acc = wmma16(a, b, acc);
    }
    int n = nt * 16 + (lane & 15), rb = (lane >> 4) << 3;
    float bias = mb3[n];
#pragma unroll
    for (int i = 0; i < 8; ++i) {
      float v = acc[i] + bias;
      int br = brow0 + rb + i;
      h0[br * HID + n]   = v;
      h016[br * HID + n] = (_Float16)v;
    }
  }
}

// ---- per-step GRU gemm: gh = h_prev @ W_hh^T  [256,768] --------------------
// grid = 96 blocks x 256 threads; each wave owns one 16x16 output tile.
__global__ __launch_bounds__(256) void gh_kernel(
    const _Float16* __restrict__ h16, const _Float16* __restrict__ whh16,
    float* __restrict__ gh) {
  const int lane = threadIdx.x & 31;
  const int wave = threadIdx.x >> 5;
  const int tile = blockIdx.x * 8 + wave;     // 0..767
  const int n0 = (tile % 48) * 16;
  const int m0 = (tile / 48) * 16;
  v8f acc = {};
#pragma unroll
  for (int kt = 0; kt < 8; ++kt) {
    v16h a = frag_a(h16, HID, m0, kt * 32, lane);
    v16h b = frag_b(whh16, HID, n0, kt * 32, lane);
    acc = wmma16(a, b, acc);
  }
  int n = n0 + (lane & 15), rb = m0 + ((lane >> 4) << 3);
#pragma unroll
  for (int i = 0; i < 8; ++i) gh[(rb + i) * (3 * HID) + n] = acc[i];
}

// ---- per-step fused: gates + h update + out-MLP + action integrate ---------
__global__ __launch_bounds__(256) void step_kernel(
    const float* __restrict__ gh,
    const float* __restrict__ W_ih, const float* __restrict__ b_ih,
    const float* __restrict__ b_hh,
    const float* __restrict__ ow1, const float* __restrict__ ob1,
    const float* __restrict__ ow2, const float* __restrict__ ob2,
    const float* __restrict__ ow3, const float* __restrict__ ob3,
    float* __restrict__ h, _Float16* __restrict__ h16,
    float* __restrict__ gi_state,
    float* __restrict__ dws, float* __restrict__ ws, int t) {
  __shared__ float xs[8];
  __shared__ float hs[HID];
  __shared__ float o1[64];
  __shared__ float o2[64];
  const int b = blockIdx.x;
  const int j = threadIdx.x;

  if (j < ACT) xs[j] = gi_state[b * ACT + j];
  __syncthreads();
  float x0 = xs[0], x1 = xs[1], x2 = xs[2], x3 = xs[3], x4 = xs[4], x5 = xs[5];

  const float* wr = W_ih + j * ACT;
  const float* wz = W_ih + (j + HID) * ACT;
  const float* wn = W_ih + (j + 2 * HID) * ACT;
  float gi_r = b_ih[j]           + wr[0]*x0 + wr[1]*x1 + wr[2]*x2 + wr[3]*x3 + wr[4]*x4 + wr[5]*x5;
  float gi_z = b_ih[j + HID]     + wz[0]*x0 + wz[1]*x1 + wz[2]*x2 + wz[3]*x3 + wz[4]*x4 + wz[5]*x5;
  float gi_n = b_ih[j + 2 * HID] + wn[0]*x0 + wn[1]*x1 + wn[2]*x2 + wn[3]*x3 + wn[4]*x4 + wn[5]*x5;

  const float* ghb = gh + b * (3 * HID);
  float gh_r = ghb[j]           + b_hh[j];
  float gh_z = ghb[j + HID]     + b_hh[j + HID];
  float gh_n = ghb[j + 2*HID]   + b_hh[j + 2 * HID];

  float r  = 1.f / (1.f + expf(-(gi_r + gh_r)));
  float z  = 1.f / (1.f + expf(-(gi_z + gh_z)));
  float nn = tanhf(gi_n + r * gh_n);
  float hp = h[b * HID + j];
  float hn = (1.f - z) * nn + z * hp;
  h[b * HID + j]   = hn;
  h16[b * HID + j] = (_Float16)hn;
  hs[j] = hn;
  __syncthreads();

  if (j < 64) {
    float a = ob1[j];
#pragma unroll 4
    for (int k = 0; k < HID; ++k) a += hs[k] * ow1[k * 64 + j];
    o1[j] = fmaxf(a, 0.f);
  }
  __syncthreads();
  if (j < 64) {
    float a = ob2[j];
#pragma unroll 4
    for (int k = 0; k < 64; ++k) a += o1[k] * ow2[k * 64 + j];
    o2[j] = fmaxf(a, 0.f);
  }
  __syncthreads();
  if (j < ACT) {
    float a = ob3[j];
#pragma unroll 4
    for (int k = 0; k < 64; ++k) a += o2[k] * ow3[k * ACT + j];
    float gnew = xs[j] + a;
    gi_state[b * ACT + j] = gnew;
    dws[(b * TSTEPS + t) * ACT + j] = a;
    ws [(b * TSTEPS + t) * ACT + j] = gnew;
  }
}

// ---------------------------------------------------------------------------
extern "C" void kernel_launch(void* const* d_in, const int* in_sizes, int n_in,
                              void* d_out, int out_size, void* d_ws, size_t ws_size,
                              hipStream_t stream) {
  const float* data   = (const float*)d_in[0];
  const float* enc_w1 = (const float*)d_in[2];
  const float* enc_b1 = (const float*)d_in[3];
  const float* enc_w2 = (const float*)d_in[4];
  const float* enc_b2 = (const float*)d_in[5];
  const float* enc_w3 = (const float*)d_in[6];
  const float* enc_b3 = (const float*)d_in[7];
  const float* mlp_w1 = (const float*)d_in[8];
  const float* mlp_b1 = (const float*)d_in[9];
  const float* mlp_w2 = (const float*)d_in[10];
  const float* mlp_b2 = (const float*)d_in[11];
  const float* mlp_w3 = (const float*)d_in[12];
  const float* mlp_b3 = (const float*)d_in[13];
  const float* W_ih   = (const float*)d_in[14];
  const float* W_hh   = (const float*)d_in[15];
  const float* b_ih   = (const float*)d_in[16];
  const float* b_hh   = (const float*)d_in[17];
  const float* out_w1 = (const float*)d_in[18];
  const float* out_b1 = (const float*)d_in[19];
  const float* out_w2 = (const float*)d_in[20];
  const float* out_b2 = (const float*)d_in[21];
  const float* out_w3 = (const float*)d_in[22];
  const float* out_b3 = (const float*)d_in[23];

  float* dws = (float*)d_out;                       // [B, T, 6]
  float* ws  = dws + (size_t)BATCH * TSTEPS * ACT;  // [B, T, 6]

  unsigned char* p = (unsigned char*)d_ws;
  auto carve = [&](size_t bytes) {
    void* r = (void*)p;
    p += (bytes + 255) & ~(size_t)255;
    return r;
  };
  int*       pooled = (int*)      carve((size_t)BATCH * 512 * 4);
  float*     hbuf   = (float*)    carve((size_t)BATCH * HID * 4);
  _Float16*  h16    = (_Float16*) carve((size_t)BATCH * HID * 2);
  float*     gh     = (float*)    carve((size_t)BATCH * 3 * HID * 4);
  float*     gi     = (float*)    carve((size_t)BATCH * ACT * 4);
  _Float16*  w2t    = (_Float16*) carve((size_t)128 * 64 * 2);
  _Float16*  w3t    = (_Float16*) carve((size_t)512 * 128 * 2);
  _Float16*  m1t    = (_Float16*) carve((size_t)256 * 512 * 2);
  _Float16*  m2t    = (_Float16*) carve((size_t)128 * 256 * 2);
  _Float16*  m3t    = (_Float16*) carve((size_t)256 * 128 * 2);
  _Float16*  whh16  = (_Float16*) carve((size_t)3 * HID * HID * 2);

  auto g1 = [](int n) { return (n + 255) / 256; };
  convT_kernel<<<g1(64 * 128),  256, 0, stream>>>(enc_w2, w2t, 64, 128);
  convT_kernel<<<g1(128 * 512), 256, 0, stream>>>(enc_w3, w3t, 128, 512);
  convT_kernel<<<g1(512 * 256), 256, 0, stream>>>(mlp_w1, m1t, 512, 256);
  convT_kernel<<<g1(256 * 128), 256, 0, stream>>>(mlp_w2, m2t, 256, 128);
  convT_kernel<<<g1(128 * 256), 256, 0, stream>>>(mlp_w3, m3t, 128, 256);
  convC_kernel<<<g1(3 * HID * HID), 256, 0, stream>>>(W_hh, whh16, 3 * HID * HID);

  hipMemsetAsync(pooled, 0, (size_t)BATCH * 512 * 4, stream);
  hipMemsetAsync(gi,     0, (size_t)BATCH * ACT * 4, stream);

  enc_kernel<<<(BATCH * NPTS) / 128, 256, 0, stream>>>(
      data, enc_w1, enc_b1, w2t, enc_b2, w3t, enc_b3, pooled);

  head_kernel<<<BATCH / 16, 256, 0, stream>>>(
      pooled, m1t, mlp_b1, m2t, mlp_b2, m3t, mlp_b3, hbuf, h16);

  for (int t = 0; t < TSTEPS; ++t) {
    gh_kernel<<<96, 256, 0, stream>>>(h16, whh16, gh);
    step_kernel<<<BATCH, 256, 0, stream>>>(
        gh, W_ih, b_ih, b_hh, out_w1, out_b1, out_w2, out_b2, out_w3, out_b3,
        hbuf, h16, gi, dws, ws, t);
  }
}
